// Next_Node_Probability_Calculator_for_group_28097676051164
// MI455X (gfx1250) — matrix operand order repaired and checked by reading, not compiled, not run on hardware
//
#include <hip/hip_runtime.h>
#include <hip/hip_bf16.h>

// Fused pointer-attention for MI455X (gfx1250, wave32, WMMA).
// One workgroup (256 thr = 8 waves) per batch element. All per-batch tensors
// staged in LDS as f16; all GEMMs via v_wmma_f32_16x16x32_f16 (f32 accum).
// G,P padded 100->112 (7 tiles of 16); E=128 (4 K-steps of 32).

typedef __attribute__((ext_vector_type(16))) _Float16 v16h;
typedef __attribute__((ext_vector_type(8)))  float    v8f;

#define LDH 136   // f16 row stride (128 + 8 pad -> 272B, avoids 64-bank conflicts)

__device__ __forceinline__ v8f wmma16(v16h a, v16h b, v8f c) {
  // (neg_a, A, neg_b, B, c_mod, C, reuse_a, reuse_b)
  return __builtin_amdgcn_wmma_f32_16x16x32_f16(false, a, false, b, (short)0, c, false, false);
}

// A-fragment 16x32 f16 from row-major LDS. k order used here (B frags match):
// lanes 0-15: M=lane, element h -> K = h; lanes 16-31: K = 16 + h.
__device__ __forceinline__ v16h a_frag_lds(const _Float16* base, int ld, int m0, int k0, int kmax) {
  const int lane = threadIdx.x & 31;
  const int l = lane & 15, hi = lane >> 4;
  const _Float16* row = base + (m0 + l) * ld + k0;
  v16h a;
#pragma unroll
  for (int h = 0; h < 16; ++h) {
    int k = hi * 16 + h;
    a[h] = (k < kmax) ? row[k] : (_Float16)0.0f;
  }
  return a;
}

// B-fragment 32x16: lanes 0-15 hold K=0..15 (elem h -> K=h), lanes 16-31 K=16..31; N=lane&15.
// Row-major f16 LDS source: B[k][n] = base[(k0+k)*ld + n0+n]; rows >= kguard read as 0.
__device__ __forceinline__ v16h b_frag_lds_rm(const _Float16* base, int ld, int k0, int n0, int kguard) {
  const int lane = threadIdx.x & 31;
  const int n = lane & 15, hi = lane >> 4;
  v16h b;
#pragma unroll
  for (int h = 0; h < 16; ++h) {
    int k = k0 + hi * 16 + h;
    b[h] = (k < kguard) ? base[k * ld + n0 + n] : (_Float16)0.0f;
  }
  return b;
}

// Transposed B source (B = Xᵀ): B[k][n] = base[(n0+n)*ld + c0 + k]; k >= kmax -> 0.
__device__ __forceinline__ v16h b_frag_lds_tr(const _Float16* base, int ld, int n0, int c0, int kmax) {
  const int lane = threadIdx.x & 31;
  const int n = lane & 15, hi = lane >> 4;
  const _Float16* row = base + (n0 + n) * ld + c0;
  v16h b;
#pragma unroll
  for (int h = 0; h < 16; ++h) {
    int k = hi * 16 + h;
    b[h] = (k < kmax) ? row[k] : (_Float16)0.0f;
  }
  return b;
}

// B-fragment from row-major f32 global weight matrix (L2-resident), converted to f16.
__device__ __forceinline__ v16h b_frag_gl(const float* __restrict__ W, int ld, int k0, int n0) {
  const int lane = threadIdx.x & 31;
  const int n = lane & 15, hi = lane >> 4;
  v16h b;
#pragma unroll
  for (int h = 0; h < 16; ++h) {
    int k = k0 + hi * 16 + h;
    b[h] = (_Float16)W[k * ld + n0 + n];
  }
  return b;
}

// C-fragment 16x16 f32: VGPR r: lanes 0-15 -> (M=r, N=lane), lanes 16-31 -> (M=r+8, N=lane-16).
__device__ __forceinline__ void c_store_lds_f16(const v8f& c, _Float16* base, int ld, int m0, int n0) {
  const int lane = threadIdx.x & 31;
  const int n = lane & 15, hi = lane >> 4;
#pragma unroll
  for (int r = 0; r < 8; ++r)
    base[(m0 + hi * 8 + r) * ld + n0 + n] = (_Float16)c[r];
}

__global__ __launch_bounds__(256, 1)
void pointer_attn_fused(const float* __restrict__ in1,   // [B,1,128]
                        const float* __restrict__ in2,   // [B,100,128]
                        const float* __restrict__ rem,   // [B,100,1]
                        const float* __restrict__ mask,  // [B,100,100]
                        const float* __restrict__ enc,   // [B,100,128]
                        const float* __restrict__ Wq,    // [257,128]
                        const float* __restrict__ Wk,    // [128,128]
                        const float* __restrict__ Wv,    // [128,128]
                        const float* __restrict__ Wc,    // [128,128]
                        const float* __restrict__ bc,    // [128]
                        float* __restrict__ out)         // [B,100,100]
{
  __shared__ _Float16 s_enc[112 * LDH];   // encoded_nodes (f16)
  __shared__ _Float16 s_a  [112 * LDH];   // input2, reused as out_concat
  __shared__ _Float16 s_q  [112 * LDH];   // Q, reused as mh_atten_out
  __shared__ _Float16 s_k  [112 * LDH];   // K
  __shared__ _Float16 s_v  [112 * LDH];   // V
  __shared__ _Float16 s_w  [8 * 16 * LDH]; // per-wave attention-weight scratch
  __shared__ float s_qbase[128];
  __shared__ float s_wqlast[128];
  __shared__ float s_bc[128];

  const int b    = blockIdx.x;
  const int tid  = threadIdx.x;
  const int wave = tid >> 5;
  const int lane = tid & 31;
  const int ncol = lane & 15;
  const int hi   = lane >> 4;

  // ---------------- Stage 0: stage per-batch tensors into LDS (f16) -------------
  for (int i = tid; i < 112 * 128; i += 256) {
    int r = i >> 7, c = i & 127;
    float ev = (r < 100) ? enc[((size_t)b * 100 + r) * 128 + c] : 0.0f;
    float iv = (r < 100) ? in2[((size_t)b * 100 + r) * 128 + c] : 0.0f;
    s_enc[r * LDH + c] = (_Float16)ev;
    s_a [r * LDH + c] = (_Float16)iv;
  }
  if (tid < 128) {
    // rank-1 part of Q: in1 . Wq[0:128, n]  (f32, shared by all g)
    float s = 0.0f;
    const float* i1 = in1 + (size_t)b * 128;
    for (int e = 0; e < 128; ++e) s += i1[e] * Wq[e * 128 + tid];
    s_qbase[tid]  = s;
    s_wqlast[tid] = Wq[256 * 128 + tid];  // remaining_loaded row of Wq
    s_bc[tid]     = bc[tid];
  }
  __syncthreads();

  // ---------------- Stage 1: K = enc@Wk, V = enc@Wv, Q = in2@Wq_mid + base -----
  {
    const int n0 = wave * 16;                 // wave -> one of 8 N-tiles
    v16h bk[4], bv[4], bq[4];
#pragma unroll
    for (int ks = 0; ks < 4; ++ks) {
      bk[ks] = b_frag_gl(Wk, 128, ks * 32, n0);
      bv[ks] = b_frag_gl(Wv, 128, ks * 32, n0);
      bq[ks] = b_frag_gl(Wq + 128 * 128, 128, ks * 32, n0);  // rows 128..255
    }
    for (int mt = 0; mt < 7; ++mt) {
      v8f ck = {}, cv = {}, cq = {};
#pragma unroll
      for (int ks = 0; ks < 4; ++ks) {
        v16h ae = a_frag_lds(s_enc, LDH, mt * 16, ks * 32, 32);
        v16h ai = a_frag_lds(s_a , LDH, mt * 16, ks * 32, 32);
        ck = wmma16(ae, bk[ks], ck);
        cv = wmma16(ae, bv[ks], cv);
        cq = wmma16(ai, bq[ks], cq);
      }
      c_store_lds_f16(ck, s_k, LDH, mt * 16, n0);
      c_store_lds_f16(cv, s_v, LDH, mt * 16, n0);
      const float qb = s_qbase[n0 + ncol];
      const float wl = s_wqlast[n0 + ncol];
#pragma unroll
      for (int r = 0; r < 8; ++r) {
        int g = mt * 16 + hi * 8 + r;
        float rv = (g < 100) ? rem[(size_t)b * 100 + g] : 0.0f;
        cq[r] = cq[r] + qb + rv * wl;
      }
      c_store_lds_f16(cq, s_q, LDH, mt * 16, n0);
    }
  }
  __syncthreads();

  // ---------------- Stage 2: multi-head attention, unit = (head, g-tile) -------
  _Float16* ws = s_w + wave * 16 * LDH;
  for (int u = wave; u < 56; u += 8) {
    const int h  = u / 7;
    const int gt = u % 7;
    // scores: Q_h (16x16, padded K=32) x K_hᵀ  -> 7 P-tiles
    v16h aq = a_frag_lds(s_q, LDH, gt * 16, h * 16, 16);
    v8f sc[7];
#pragma unroll
    for (int pt = 0; pt < 7; ++pt) {
      v16h bkT = b_frag_lds_tr(s_k, LDH, pt * 16, h * 16, 16);
      v8f z = {};
      sc[pt] = wmma16(aq, bkT, z);
    }
    // scale (1/sqrt(16)) + additive mask; invalid p -> -inf
#pragma unroll
    for (int pt = 0; pt < 7; ++pt) {
      int p = pt * 16 + ncol;
      bool pv = (p < 100);
#pragma unroll
      for (int r = 0; r < 8; ++r) {
        int g = gt * 16 + hi * 8 + r;
        float v = sc[pt][r] * 0.25f;
        if (pv) v += (g < 100) ? mask[((size_t)b * 100 + g) * 100 + p] : 0.0f;
        else    v = -INFINITY;
        sc[pt][r] = v;
      }
    }
    // softmax over P: reduce across 7 tiles in-register, then 16-lane butterfly
#pragma unroll
    for (int r = 0; r < 8; ++r) {
      float m = -INFINITY;
#pragma unroll
      for (int pt = 0; pt < 7; ++pt) m = fmaxf(m, sc[pt][r]);
#pragma unroll
      for (int off = 1; off < 16; off <<= 1) m = fmaxf(m, __shfl_xor(m, off, 32));
      float s = 0.0f;
#pragma unroll
      for (int pt = 0; pt < 7; ++pt) { float e = __expf(sc[pt][r] - m); sc[pt][r] = e; s += e; }
#pragma unroll
      for (int off = 1; off < 16; off <<= 1) s += __shfl_xor(s, off, 32);
      float inv = 1.0f / s;
#pragma unroll
      for (int pt = 0; pt < 7; ++pt) sc[pt][r] *= inv;
    }
    // re-layout weights as A-matrix via per-wave LDS scratch (cols 112..127 zeroed)
#pragma unroll
    for (int pt = 0; pt < 7; ++pt) c_store_lds_f16(sc[pt], ws, LDH, 0, pt * 16);
#pragma unroll
    for (int r = 0; r < 8; ++r) ws[(hi * 8 + r) * LDH + 112 + ncol] = (_Float16)0.0f;
    // out_h = weights (16x128p) @ V_h (128p x16); guard V rows >= 112
    v8f co = {};
#pragma unroll
    for (int ks = 0; ks < 4; ++ks) {
      v16h aw  = a_frag_lds(ws, LDH, 0, ks * 32, 32);
      v16h bvh = b_frag_lds_rm(s_v, LDH, ks * 32, h * 16, 112);
      co = wmma16(aw, bvh, co);
    }
    c_store_lds_f16(co, s_a, LDH, gt * 16, h * 16);   // out_concat (reuses in2 buf)
  }
  __syncthreads();

  // ---------------- Stage 3: mh = out_concat @ Wc + bc -------------------------
  {
    const int n0 = wave * 16;
    v16h bw[4];
#pragma unroll
    for (int ks = 0; ks < 4; ++ks) bw[ks] = b_frag_gl(Wc, 128, ks * 32, n0);
    const float bias = s_bc[n0 + ncol];
    for (int mt = 0; mt < 7; ++mt) {
      v8f c = {};
#pragma unroll
      for (int ks = 0; ks < 4; ++ks) {
        v16h a = a_frag_lds(s_a, LDH, mt * 16, ks * 32, 32);
        c = wmma16(a, bw[ks], c);
      }
#pragma unroll
      for (int r = 0; r < 8; ++r) c[r] += bias;
      c_store_lds_f16(c, s_q, LDH, mt * 16, n0);      // mh_atten_out (reuses Q buf)
    }
  }
  __syncthreads();

  // ---------------- Stage 4: pointer scores + tanh clip + softmax -> probs -----
  if (wave < 7) {
    const int gt = wave;
    v8f sc[7];
#pragma unroll
    for (int pt = 0; pt < 7; ++pt) { v8f z = {}; sc[pt] = z; }
    for (int ks = 0; ks < 4; ++ks) {
      v16h a = a_frag_lds(s_q, LDH, gt * 16, ks * 32, 32);
#pragma unroll
      for (int pt = 0; pt < 7; ++pt) {
        v16h bT = b_frag_lds_tr(s_enc, LDH, pt * 16, ks * 32, 32);
        sc[pt] = wmma16(a, bT, sc[pt]);
      }
    }
    const float invs = 0.08838834764831845f;  // 1/sqrt(128)
#pragma unroll
    for (int pt = 0; pt < 7; ++pt) {
      int p = pt * 16 + ncol;
      bool pv = (p < 100);
#pragma unroll
      for (int r = 0; r < 8; ++r) {
        int g = gt * 16 + hi * 8 + r;
        float v = 10.0f * tanhf(sc[pt][r] * invs);
        if (pv) v += (g < 100) ? mask[((size_t)b * 100 + g) * 100 + p] : 0.0f;
        else    v = -INFINITY;
        sc[pt][r] = v;
      }
    }
#pragma unroll
    for (int r = 0; r < 8; ++r) {
      float m = -INFINITY;
#pragma unroll
      for (int pt = 0; pt < 7; ++pt) m = fmaxf(m, sc[pt][r]);
#pragma unroll
      for (int off = 1; off < 16; off <<= 1) m = fmaxf(m, __shfl_xor(m, off, 32));
      float s = 0.0f;
#pragma unroll
      for (int pt = 0; pt < 7; ++pt) { float e = __expf(sc[pt][r] - m); sc[pt][r] = e; s += e; }
#pragma unroll
      for (int off = 1; off < 16; off <<= 1) s += __shfl_xor(s, off, 32);
      float inv = 1.0f / s;
      int g = gt * 16 + hi * 8 + r;
      if (g < 100) {
#pragma unroll
        for (int pt = 0; pt < 7; ++pt) {
          int p = pt * 16 + ncol;
          if (p < 100) out[((size_t)b * 100 + g) * 100 + p] = sc[pt][r] * inv;
        }
      }
    }
  }
}

extern "C" void kernel_launch(void* const* d_in, const int* in_sizes, int n_in,
                              void* d_out, int out_size, void* d_ws, size_t ws_size,
                              hipStream_t stream) {
  const float* in1  = (const float*)d_in[0];
  const float* in2  = (const float*)d_in[1];
  const float* rem  = (const float*)d_in[2];
  const float* mask = (const float*)d_in[3];
  const float* enc  = (const float*)d_in[4];
  const float* Wq   = (const float*)d_in[5];
  const float* Wk   = (const float*)d_in[6];
  const float* Wv   = (const float*)d_in[7];
  const float* Wc   = (const float*)d_in[8];
  const float* bc   = (const float*)d_in[9];
  float* out = (float*)d_out;
  (void)in_sizes; (void)n_in; (void)out_size; (void)d_ws; (void)ws_size;

  dim3 grid(512), block(256);
  hipLaunchKernelGGL(pointer_attn_fused, grid, block, 0, stream,
                     in1, in2, rem, mask, enc, Wq, Wk, Wv, Wc, bc, out);
}
